// DGCNN_79405355368881
// MI455X (gfx1250) — compile-verified, matrix operand order checked by reading, not run on hardware
//
#include <hip/hip_runtime.h>

#define KNN_K 20
#define NPTS 2048
#define BATCH 8
#define XC_LD 512
#define BN_SCALE_C 0.9999950000374997f
#define NEG_INF (-3.4e38f)

typedef _Float16 v16h __attribute__((ext_vector_type(16)));
typedef float    v8f  __attribute__((ext_vector_type(8)));

__device__ __forceinline__ float lrelu(float v) { return v > 0.f ? v : 0.2f * v; }

__device__ __forceinline__ void atomicMaxFloat(float* addr, float val) {
    unsigned int* ai = (unsigned int*)addr;
    unsigned int old = *ai;
    while (__uint_as_float(old) < val) {
        unsigned int assumed = old;
        old = atomicCAS(ai, assumed, __float_as_uint(val));
        if (old == assumed) break;
    }
}

// Fragment permutation for 16-bit WMMA operands (ISA 7.12.2, 16-bit A 16x32):
// within a 32-wide K chunk, source position kb maps to fragment position
// half*16 + 2*i + j so a lane's 16 halfs are contiguous (2x b128 loads).
__device__ __forceinline__ int fragpos(int kb) {
    int j   = kb & 1;
    int h   = (kb >> 3) & 1;
    int rem = kb - 8 * h - j;                 // 0,2,4,6,16,18,20,22
    int i   = (rem < 8) ? (rem >> 1) : (4 + ((rem - 16) >> 1));
    return h * 16 + 2 * i + j;
}

union FragU { v16h h; uint4 q[2]; };

__device__ __forceinline__ v16h load_frag(const _Float16* p /*32B aligned*/) {
    FragU u;
    const uint4* q = reinterpret_cast<const uint4*>(p);
    u.q[0] = q[0];
    u.q[1] = q[1];
    return u.h;
}

// ---------------------------------------------------------------------------
// Weight pre-convert: f32 -> f16, permuted into fragment order (once per call,
// instead of once per point-block).
// ---------------------------------------------------------------------------
__global__ void convert_w_kernel(const float* W, _Float16* Wh, int total, int K2) {
    int i = blockIdx.x * 256 + threadIdx.x;
    if (i >= total) return;
    int o = i / K2, k = i % K2;
    Wh[(size_t)o * K2 + (k & ~31) + fragpos(k & 31)] = (_Float16)W[i];
}

// ---------------------------------------------------------------------------
// kNN: one block (256 threads) per point row. Each thread owns 8 candidate
// distances in registers; 20 rounds of block-wide argmax (tie -> lowest idx,
// matching jax.lax.top_k stability). pairwise = -||x[n]-x[m]||^2.
// ---------------------------------------------------------------------------
template<int C>
__global__ __launch_bounds__(256)
void knn_kernel(const float* xin, int ld, int coff, int* idx_out) {
    int row = blockIdx.x;              // b*NPTS + n
    int b   = row / NPTS;
    int t   = threadIdx.x;
    __shared__ float ctr[C];
    __shared__ float bv[256];
    __shared__ int   bi[256];
    __shared__ int   winner;
    for (int c = t; c < C; c += 256) ctr[c] = xin[(size_t)row * ld + coff + c];
    __syncthreads();

    float dist[8];
    #pragma unroll
    for (int u = 0; u < 8; ++u) {
        int m = u * 256 + t;
        const float* xr = xin + (size_t)(b * NPTS + m) * ld + coff;
        float s = 0.f;
        for (int c = 0; c < C; ++c) { float d = xr[c] - ctr[c]; s += d * d; }
        dist[u] = -s;
    }
    unsigned sel = 0;
    for (int r = 0; r < KNN_K; ++r) {
        float bvv = NEG_INF; int bii = 1 << 30;
        #pragma unroll
        for (int u = 0; u < 8; ++u) {
            if (!((sel >> u) & 1u)) {
                int m = u * 256 + t;
                if (dist[u] > bvv || (dist[u] == bvv && m < bii)) { bvv = dist[u]; bii = m; }
            }
        }
        bv[t] = bvv; bi[t] = bii;
        __syncthreads();
        for (int s = 128; s > 0; s >>= 1) {
            if (t < s) {
                if (bv[t + s] > bv[t] || (bv[t + s] == bv[t] && bi[t + s] < bi[t])) {
                    bv[t] = bv[t + s]; bi[t] = bi[t + s];
                }
            }
            __syncthreads();
        }
        if (t == 0) { winner = bi[0]; idx_out[(size_t)row * KNN_K + r] = bi[0]; }
        __syncthreads();
        int w = winner;
        if ((w & 255) == t) sel |= 1u << (w >> 8);
        __syncthreads();
    }
}

// ---------------------------------------------------------------------------
// Edge conv layer 1 (C=3 -> 64): K=6 too small for WMMA; scalar kernel.
// ---------------------------------------------------------------------------
__global__ __launch_bounds__(64)
void edge_conv1_kernel(const float* x, const int* idx, const float* W1,
                       const float* g1, const float* b1, float* xc) {
    int row = blockIdx.x; int b = row / NPTS;
    int o = threadIdx.x;
    __shared__ float ctr[3];
    __shared__ float nbr[KNN_K][3];
    if (o < 3) ctr[o] = x[(size_t)row * 3 + o];
    if (o < KNN_K) {
        int m = idx[(size_t)row * KNN_K + o];
        const float* xr = x + (size_t)(b * NPTS + m) * 3;
        for (int c = 0; c < 3; ++c) nbr[o][c] = xr[c];
    }
    __syncthreads();
    float w[6];
    #pragma unroll
    for (int c = 0; c < 6; ++c) w[c] = W1[o * 6 + c];
    float go = g1[o], bo = b1[o];
    float best = NEG_INF;
    for (int kk = 0; kk < KNN_K; ++kk) {
        float y = 0.f;
        #pragma unroll
        for (int c = 0; c < 3; ++c)
            y += w[c] * (nbr[kk][c] - ctr[c]) + w[3 + c] * ctr[c];
        best = fmaxf(best, lrelu(go * (y * BN_SCALE_C) + bo));
    }
    xc[(size_t)row * XC_LD + o] = best;
}

// ---------------------------------------------------------------------------
// Edge conv (WMMA): one block (4 waves) per point. Feat (32 x 2C, rows>=20
// zero) built in LDS in fragment-permuted f16 order; weights come from the
// pre-permuted f16 copy. Fragment loads are 2x16B vector loads per operand.
// ---------------------------------------------------------------------------
template<int CIN, int COUT>
__global__ __launch_bounds__(128)
void edge_conv_wmma_kernel(const float* xcbuf, int in_off, int out_off,
                           const int* idx, const _Float16* __restrict__ Wh,
                           const float* gg, const float* bb, float* xcout) {
    constexpr int K2 = 2 * CIN;
    int row = blockIdx.x; int b = row / NPTS;
    __shared__ __align__(16) _Float16 F[32][K2];
    __shared__ int nb[KNN_K];
    int t = threadIdx.x;
    if (t < KNN_K) nb[t] = idx[(size_t)row * KNN_K + t];
    __syncthreads();
    const float* ctr = xcbuf + (size_t)row * XC_LD + in_off;
    for (int p = t; p < 32 * K2; p += 128) {
        int kk = p / K2, c = p % K2;
        float v = 0.f;
        if (kk < KNN_K) {
            if (c < CIN) v = xcbuf[(size_t)(b * NPTS + nb[kk]) * XC_LD + in_off + c] - ctr[c];
            else         v = ctr[c - CIN];
        }
        F[kk][(c & ~31) + fragpos(c & 31)] = (_Float16)v;
    }
    __syncthreads();

    int wave = t / 32, lane = t % 32;
    int lm = lane % 16, half = lane / 16;
    for (int nt = wave; nt < COUT / 16; nt += 4) {
        int o = nt * 16 + lm;
        const _Float16* wrow = Wh + (size_t)o * K2;
        __builtin_prefetch(wrow, 0, 1);          // global_prefetch_b8
        v8f acc0 = {}; v8f acc1 = {};
        for (int kt = 0; kt < K2 / 32; ++kt) {
            int base = kt * 32 + half * 16;
            v16h a0 = load_frag(&F[lm][base]);
            v16h a1 = load_frag(&F[16 + lm][base]);
            v16h bf = load_frag(wrow + base);
            acc0 = __builtin_amdgcn_wmma_f32_16x16x32_f16(false, a0, false, bf,
                                                          (short)0, acc0, false, false);
            acc1 = __builtin_amdgcn_wmma_f32_16x16x32_f16(false, a1, false, bf,
                                                          (short)0, acc1, false, false);
        }
        float go = gg[o], bo = bb[o];
        float best = NEG_INF;
        #pragma unroll
        for (int r = 0; r < 8; ++r)       // tile0: rows 0..15 all valid
            best = fmaxf(best, lrelu(go * (acc0[r] * BN_SCALE_C) + bo));
        if (half == 0) {
            #pragma unroll
            for (int r = 0; r < 4; ++r)   // tile1: rows 16..19 valid
                best = fmaxf(best, lrelu(go * (acc1[r] * BN_SCALE_C) + bo));
        }
        best = fmaxf(best, __shfl_xor(best, 16, 32));
        if (half == 0) xcout[(size_t)row * XC_LD + out_off + o] = best;
    }
}

// ---------------------------------------------------------------------------
// h = lrelu(bn(xc @ W5^T)) fused with max/mean pooling over the 2048 points.
// 16x64 output tile per block; A-tile staged in LDS (f16, fragment order).
// ---------------------------------------------------------------------------
__global__ __launch_bounds__(128)
void w5_pool_kernel(const float* xc, const _Float16* __restrict__ Wh5,
                    const float* g5, const float* b5, float* hmax, float* hsum) {
    int row0 = blockIdx.x * 16;
    int t = threadIdx.x;
    __shared__ __align__(16) _Float16 As[16][512];
    for (int p = t; p < 16 * 512; p += 128) {
        int r = p >> 9, k = p & 511;
        As[r][(k & ~31) + fragpos(k & 31)] = (_Float16)xc[(size_t)(row0 + r) * XC_LD + k];
    }
    __syncthreads();

    int wave = t / 32, lane = t % 32;
    int lm = lane % 16, half = lane / 16;
    int o = blockIdx.y * 64 + wave * 16 + lm;
    int b = row0 / NPTS;
    const _Float16* wrow = Wh5 + (size_t)o * 512;
    v8f acc = {};
    for (int kt = 0; kt < 512 / 32; ++kt) {
        int base = kt * 32 + half * 16;
        v16h a  = load_frag(&As[lm][base]);
        v16h bf = load_frag(wrow + base);
        acc = __builtin_amdgcn_wmma_f32_16x16x32_f16(false, a, false, bf,
                                                     (short)0, acc, false, false);
    }
    float go = g5[o], bo = b5[o];
    float mx = NEG_INF, sm = 0.f;
    #pragma unroll
    for (int r = 0; r < 8; ++r) {
        float v = lrelu(go * (acc[r] * BN_SCALE_C) + bo);
        mx = fmaxf(mx, v); sm += v;
    }
    mx = fmaxf(mx, __shfl_xor(mx, 16, 32));
    sm += __shfl_xor(sm, 16, 32);
    if (half == 0) {
        atomicMaxFloat(&hmax[b * 1024 + o], mx);
        atomicAdd(&hsum[b * 1024 + o], sm);
    }
}

__global__ void init_pool_kernel(float* hmax, float* hsum) {
    int i = blockIdx.x * 256 + threadIdx.x;
    if (i < BATCH * 1024) { hmax[i] = NEG_INF; hsum[i] = 0.f; }
}

// ---------------------------------------------------------------------------
// FC head (tiny): g = [hmax, hsum/2048]; three dense layers.
// ---------------------------------------------------------------------------
__global__ void fc1_kernel(const float* hmax, const float* hsum, const float* L1,
                           const float* g6, const float* b6, float* out1) {
    int i = blockIdx.x * 256 + threadIdx.x;
    if (i >= BATCH * 512) return;
    int b = i / 512, o = i % 512;
    const float* w = L1 + (size_t)o * 2048;
    float s = 0.f;
    for (int j = 0; j < 1024; ++j) s += hmax[b * 1024 + j] * w[j];
    for (int j = 0; j < 1024; ++j) s += (hsum[b * 1024 + j] * (1.f / NPTS)) * w[1024 + j];
    out1[i] = lrelu(g6[o] * (s * BN_SCALE_C) + b6[o]);
}

__global__ void fc2_kernel(const float* in, const float* L2, const float* L2b,
                           const float* g7, const float* b7, float* out2) {
    int i = blockIdx.x * 256 + threadIdx.x;
    if (i >= BATCH * 256) return;
    int b = i / 256, o = i % 256;
    float s = L2b[o];
    const float* w = L2 + (size_t)o * 512;
    for (int j = 0; j < 512; ++j) s += in[b * 512 + j] * w[j];
    out2[i] = lrelu(g7[o] * (s * BN_SCALE_C) + b7[o]);
}

__global__ void fc3_kernel(const float* in, const float* L3, const float* L3b,
                           float* out) {
    int i = blockIdx.x * 128 + threadIdx.x;
    if (i >= BATCH * 10) return;
    int b = i / 10, o = i % 10;
    float s = L3b[o];
    const float* w = L3 + (size_t)o * 256;
    for (int j = 0; j < 256; ++j) s += in[b * 256 + j] * w[j];
    out[i] = s;
}

// ---------------------------------------------------------------------------
extern "C" void kernel_launch(void* const* d_in, const int* in_sizes, int n_in,
                              void* d_out, int out_size, void* d_ws, size_t ws_size,
                              hipStream_t stream) {
    const float* x   = (const float*)d_in[0];
    const float* W1  = (const float*)d_in[1];
    const float* g1  = (const float*)d_in[2];
    const float* b1  = (const float*)d_in[3];
    const float* W2  = (const float*)d_in[4];
    const float* g2  = (const float*)d_in[5];
    const float* b2  = (const float*)d_in[6];
    const float* W3  = (const float*)d_in[7];
    const float* g3  = (const float*)d_in[8];
    const float* b3  = (const float*)d_in[9];
    const float* W4  = (const float*)d_in[10];
    const float* g4  = (const float*)d_in[11];
    const float* b4  = (const float*)d_in[12];
    const float* W5  = (const float*)d_in[13];
    const float* g5  = (const float*)d_in[14];
    const float* b5  = (const float*)d_in[15];
    const float* L1  = (const float*)d_in[16];
    const float* g6  = (const float*)d_in[17];
    const float* b6  = (const float*)d_in[18];
    const float* L2  = (const float*)d_in[19];
    const float* L2b = (const float*)d_in[20];
    const float* g7  = (const float*)d_in[21];
    const float* b7  = (const float*)d_in[22];
    const float* L3  = (const float*)d_in[23];
    const float* L3b = (const float*)d_in[24];

    char* ws = (char*)d_ws;
    size_t off = 0;
    float* xc   = (float*)(ws + off); off += (size_t)16384 * 512 * 4;   // 32 MiB
    int*   idx  = (int*)(ws + off);   off += (size_t)16384 * KNN_K * 4; // 1.25 MiB
    float* hmax = (float*)(ws + off); off += (size_t)BATCH * 1024 * 4;
    float* hsum = (float*)(ws + off); off += (size_t)BATCH * 1024 * 4;
    float* o1   = (float*)(ws + off); off += (size_t)BATCH * 512 * 4;
    float* o2   = (float*)(ws + off); off += (size_t)BATCH * 256 * 4;
    off = (off + 255) & ~(size_t)255;
    _Float16* Wh2 = (_Float16*)(ws + off); off += (size_t)64 * 128 * 2;
    _Float16* Wh3 = (_Float16*)(ws + off); off += (size_t)128 * 128 * 2;
    _Float16* Wh4 = (_Float16*)(ws + off); off += (size_t)256 * 256 * 2;
    _Float16* Wh5 = (_Float16*)(ws + off); off += (size_t)1024 * 512 * 2;

    const int P = BATCH * NPTS;  // 16384 points

    init_pool_kernel<<<(BATCH * 1024 + 255) / 256, 256, 0, stream>>>(hmax, hsum);
    convert_w_kernel<<<(64 * 128 + 255) / 256, 256, 0, stream>>>(W2, Wh2, 64 * 128, 128);
    convert_w_kernel<<<(128 * 128 + 255) / 256, 256, 0, stream>>>(W3, Wh3, 128 * 128, 256);
    convert_w_kernel<<<(256 * 256 + 255) / 256, 256, 0, stream>>>(W4, Wh4, 256 * 256, 256);
    convert_w_kernel<<<(1024 * 512 + 255) / 256, 256, 0, stream>>>(W5, Wh5, 1024 * 512, 512);

    // Layer 1
    knn_kernel<3><<<P, 256, 0, stream>>>(x, 3, 0, idx);
    edge_conv1_kernel<<<P, 64, 0, stream>>>(x, idx, W1, g1, b1, xc);
    // Layer 2: in xc[:,0:64) -> out xc[:,64:128)
    knn_kernel<64><<<P, 256, 0, stream>>>(xc, XC_LD, 0, idx);
    edge_conv_wmma_kernel<64, 64><<<P, 128, 0, stream>>>(xc, 0, 64, idx, Wh2, g2, b2, xc);
    // Layer 3: in xc[:,64:128) -> out xc[:,128:256)
    knn_kernel<64><<<P, 256, 0, stream>>>(xc, XC_LD, 64, idx);
    edge_conv_wmma_kernel<64, 128><<<P, 128, 0, stream>>>(xc, 64, 128, idx, Wh3, g3, b3, xc);
    // Layer 4: in xc[:,128:256) -> out xc[:,256:512)
    knn_kernel<128><<<P, 256, 0, stream>>>(xc, XC_LD, 128, idx);
    edge_conv_wmma_kernel<128, 256><<<P, 128, 0, stream>>>(xc, 128, 256, idx, Wh4, g4, b4, xc);

    // MLP + fused pooling
    dim3 g5grid(P / 16, 1024 / 64);
    w5_pool_kernel<<<g5grid, 128, 0, stream>>>(xc, Wh5, g5, b5, hmax, hsum);

    fc1_kernel<<<(BATCH * 512 + 255) / 256, 256, 0, stream>>>(hmax, hsum, L1, g6, b6, o1);
    fc2_kernel<<<(BATCH * 256 + 255) / 256, 256, 0, stream>>>(o1, L2, L2b, g7, b7, o2);
    fc3_kernel<<<1, 128, 0, stream>>>(o2, L3, L3b, (float*)d_out);
}
